// CRF_46729244180559
// MI455X (gfx1250) — matrix operand order repaired and checked by reading, not compiled, not run on hardware
//
#include <hip/hip_runtime.h>
#include <hip/hip_bf16.h>

typedef __attribute__((ext_vector_type(16))) _Float16 v16h;
typedef __attribute__((ext_vector_type(8)))  float    v8f;
typedef __attribute__((ext_vector_type(4)))  float    v4f;

#define NLAB   26
#define NDIM   128
#define NTIME  64
#define NBATCH 8192
#define NEGINF (-1e30f)

// One workgroup = 8 waves = 8 batch elements. Each wave:
//   repeat 4x: WMMA a 16(time) x 32(label,padded) emission tile (K=128, f16 in / f32 acc)
//              -> LDS slab -> 16 forward-DP steps (logsumexp over labels in lanes).
// Gold path score gathered by lane 0 from the same LDS slab. Deterministic:
// per-wave partials summed in fixed order, block partial -> d_ws.
__global__ __launch_bounds__(256) void crf_fused_kernel(
    const float* __restrict__ feat, const int* __restrict__ yin,
    const float* __restrict__ params, float* __restrict__ partial)
{
  __shared__ __align__(32) _Float16 Wh[32][NDIM];   // 8 KB, labels 26..31 zero
  __shared__ float emb[8][16][32];                  // 16 KB, per-wave emission slabs
  __shared__ float wavePart[8];

  const int tid  = threadIdx.x;
  const int wave = tid >> 5;
  const int lane = tid & 31;

  // Stage emission weights as f16 (zero-padded to 32 labels).
  for (int i = tid; i < 32 * NDIM; i += 256) {
    const int l = i >> 7, k = i & (NDIM - 1);
    Wh[l][k] = (_Float16)((l < NLAB) ? params[l * NDIM + k] : 0.0f);
  }
  __syncthreads();

  const int b = blockIdx.x * 8 + wave;
  const float* __restrict__ frow = feat + (size_t)b * NTIME * NDIM;
  const int*   __restrict__ yb   = yin + b * NTIME;
  const float* __restrict__ Tr   = params + NLAB * NDIM;

  // Transition column for this lane's label: TrCol[i] = Tr[i][lane]
  float TrCol[NLAB];
  #pragma unroll
  for (int i = 0; i < NLAB; ++i)
    TrCol[i] = (lane < NLAB) ? Tr[i * NLAB + lane] : 0.0f;

  const int m16 = lane & 15;   // row / column within 16-wide tile
  const int g   = lane >> 4;   // half-wave group (selects K sub-block)

  // Preload B fragments (weights) for both label tiles, all 4 K-chunks.
  // B 32x16 f16 layout: lane n = lane&15; K runs {g*8..g*8+7} and {16+g*8..16+g*8+7}
  v16h Bf[2][4];
  #pragma unroll
  for (int ct = 0; ct < 2; ++ct)
    #pragma unroll
    for (int kk = 0; kk < 4; ++kk)
      Bf[ct][kk] = __builtin_shufflevector(
          *(const __attribute__((ext_vector_type(8))) _Float16*)&Wh[ct * 16 + m16][kk * 32 + g * 8],
          *(const __attribute__((ext_vector_type(8))) _Float16*)&Wh[ct * 16 + m16][kk * 32 + 16 + g * 8],
          0, 1, 2, 3, 4, 5, 6, 7, 8, 9, 10, 11, 12, 13, 14, 15);

  float alpha    = NEGINF;
  float em_score = 0.0f, tr_score = 0.0f;
  int   yprev    = 0;

  #pragma unroll 1
  for (int chunk = 0; chunk < 4; ++chunk) {
    // ---- emission tile: 16 timesteps x 32 labels via 8 WMMAs ----
    v8f acc0 = {}; v8f acc1 = {};
    #pragma unroll
    for (int kk = 0; kk < 4; ++kk) {
      // A 16x32 f16 layout: lane row = lane&15; K runs {g*8..g*8+7} and {16+g*8..16+g*8+7}
      const float* r = frow + (size_t)(chunk * 16 + m16) * NDIM + kk * 32 + g * 8;
      v4f f0 = *(const v4f*)(r);
      v4f f1 = *(const v4f*)(r + 4);
      v4f f2 = *(const v4f*)(r + 16);
      v4f f3 = *(const v4f*)(r + 20);
      v16h a;
      #pragma unroll
      for (int j = 0; j < 4; ++j) {
        a[j]      = (_Float16)f0[j];
        a[4 + j]  = (_Float16)f1[j];
        a[8 + j]  = (_Float16)f2[j];
        a[12 + j] = (_Float16)f3[j];
      }
      acc0 = __builtin_amdgcn_wmma_f32_16x16x32_f16(false, a, false, Bf[0][kk],
                                                    (short)0, acc0, false, false);
      acc1 = __builtin_amdgcn_wmma_f32_16x16x32_f16(false, a, false, Bf[1][kk],
                                                    (short)0, acc1, false, false);
    }
    // prefetch next chunk's feature rows while we do the DP
    if (chunk < 3)
      __builtin_prefetch(frow + (size_t)((chunk + 1) * 16 + m16) * NDIM + g * 8, 0, 0);

    // D layout: lane n = lane&15, VGPR r holds M = g*8 + r  -> emb[time][label]
    #pragma unroll
    for (int rr = 0; rr < 8; ++rr) {
      emb[wave][g * 8 + rr][m16]      = acc0[rr];
      emb[wave][g * 8 + rr][16 + m16] = acc1[rr];
    }

    // ---- forward DP over these 16 timesteps (lane = label) ----
    #pragma unroll 1
    for (int tt = 0; tt < 16; ++tt) {
      const int   t = chunk * 16 + tt;
      const float e = emb[wave][tt][lane];
      if (t == 0) {
        alpha = (lane < NLAB) ? e : NEGINF;
      } else {
        float s[NLAB];
        float mx = NEGINF;
        #pragma unroll
        for (int l = 0; l < NLAB; ++l) {
          const float al = __shfl(alpha, l, 32);
          s[l] = al + TrCol[l];
          mx = fmaxf(mx, s[l]);
        }
        float sum = 0.0f;
        #pragma unroll
        for (int l = 0; l < NLAB; ++l) sum += __expf(s[l] - mx);
        const float na = mx + __logf(sum) + e;
        alpha = (lane < NLAB) ? na : NEGINF;
      }
      if (lane == 0) {  // gold path score
        const int yt = yb[t];
        em_score += emb[wave][tt][yt];
        if (t > 0) tr_score += Tr[yprev * NLAB + yt];
        yprev = yt;
      }
    }
  }

  // logZ = logsumexp over labels (held one per lane)
  float mx = alpha;
  #pragma unroll
  for (int off = 16; off > 0; off >>= 1) mx = fmaxf(mx, __shfl_xor(mx, off, 32));
  float ex = (lane < NLAB) ? __expf(alpha - mx) : 0.0f;
  #pragma unroll
  for (int off = 16; off > 0; off >>= 1) ex += __shfl_xor(ex, off, 32);
  const float logZ = mx + __logf(ex);

  if (lane == 0) wavePart[wave] = em_score + tr_score - logZ;
  __syncthreads();
  if (tid == 0) {
    float ssum = 0.0f;
    #pragma unroll
    for (int w = 0; w < 8; ++w) ssum += wavePart[w];
    partial[blockIdx.x] = ssum;
  }
}

// Fixed-order reduction of 1024 block partials -> -mean
__global__ __launch_bounds__(256) void crf_reduce_kernel(
    const float* __restrict__ partial, float* __restrict__ out)
{
  __shared__ float sm[256];
  float s = 0.0f;
  for (int i = threadIdx.x; i < 1024; i += 256) s += partial[i];
  sm[threadIdx.x] = s;
  __syncthreads();
  for (int off = 128; off > 0; off >>= 1) {
    if ((int)threadIdx.x < off) sm[threadIdx.x] += sm[threadIdx.x + off];
    __syncthreads();
  }
  if (threadIdx.x == 0) out[0] = -sm[0] / (float)NBATCH;
}

extern "C" void kernel_launch(void* const* d_in, const int* in_sizes, int n_in,
                              void* d_out, int out_size, void* d_ws, size_t ws_size,
                              hipStream_t stream) {
  const float* feat   = (const float*)d_in[0];  // [8192,64,128] f32
  const int*   y      = (const int*)d_in[1];    // [8192,64] i32
  const float* params = (const float*)d_in[2];  // [26*128 + 26*26] f32
  float* out     = (float*)d_out;
  float* partial = (float*)d_ws;                // 1024 floats

  crf_fused_kernel<<<NBATCH / 8, 256, 0, stream>>>(feat, y, params, partial);
  crf_reduce_kernel<<<1, 256, 0, stream>>>(partial, out);
}
